// ParallelFLAAttention_29403346108707
// MI455X (gfx1250) — matrix operand order repaired and checked by reading, not compiled, not run on hardware
//
#include <hip/hip_runtime.h>

// ParallelFLAAttention for MI455X (gfx1250, wave32, WMMA, async->LDS, TDM).
//   1) f32->f16 converts for X, Wq, Wk, Wv, Wo
//   2) double-buffered WMMA GEMM (async A tiles, pipelined B tiles)
//   3) row-softmax(q), row-softmax(k), gate = logsigmoid(k@Wg+bg)/16
//   4) delta-rule scan: S column-per-lane, register-resident (128 VGPRs)
//   5) causal flash attention with WMMA; K tiles via Tensor Data Mover,
//      Q tiles via GLOBAL_LOAD_ASYNC_TO_LDS_B128
//   6) WMMA GEMM epilogue: out = 0.99*(base@Wo) + 0.01*fla

typedef _Float16 half_t;
typedef __attribute__((ext_vector_type(16))) _Float16 v16h;
typedef __attribute__((ext_vector_type(8)))  float    v8f;
typedef __attribute__((ext_vector_type(4)))  unsigned int u32x4;
typedef __attribute__((ext_vector_type(8)))  int      i32x8;
typedef __attribute__((ext_vector_type(4)))  int      i32x4;

#define B_   2
#define N_   2048
#define H_   16
#define HKV_ 8
#define D_   128
#define HID_ 2048
#define SCALE_ 0.08838834764831845f   // 1/sqrt(128)

union F16x16 { v16h v; half_t h[16]; };
union Facc   { v8f  v; float  f[8];  };

__device__ __forceinline__ v8f wmma_f16(v16h a, v16h b, v8f c) {
  return __builtin_amdgcn_wmma_f32_16x16x32_f16(false, a, false, b, (short)0, c, false, false);
}

// Per-lane async 16-byte copy global -> LDS (ASYNCcnt path).
__device__ __forceinline__ void async_g2lds_b128(const void* gptr, void* lds_ptr) {
  uint32_t loff  = (uint32_t)(uintptr_t)lds_ptr;   // LDS aperture: offset = addr[31:0]
  uint64_t gaddr = (uint64_t)(uintptr_t)gptr;
  asm volatile("global_load_async_to_lds_b128 %0, %1, off"
               :: "v"(loff), "v"(gaddr) : "memory");
}
__device__ __forceinline__ void wait_async_le2() {
  asm volatile("s_wait_asynccnt 0x2" ::: "memory");
}
__device__ __forceinline__ void wait_async0() {
  asm volatile("s_wait_asynccnt 0x0" ::: "memory");
}

// TDM 2D tile load global -> LDS (TENSORcnt path). data_size = 8 bytes.
// tile_bytes_x: bytes per tile row (multiple of 8); rows: tile rows;
// row_stride_bytes: stride between tensor rows (multiple of 8).
__device__ __forceinline__ void tdm_load_2d(void* lds_ptr, const void* gptr,
                                            uint32_t tile_bytes_x, uint32_t rows,
                                            uint32_t row_stride_bytes) {
  uint32_t lds = (uint32_t)(uintptr_t)lds_ptr;
  uint64_t ga  = (uint64_t)(uintptr_t)gptr;
  uint32_t tdx = tile_bytes_x >> 3;       // dim0 length / tile_dim0 (8B units)
  uint32_t str = row_stride_bytes >> 3;   // dim0 stride (8B units)
  u32x4 g0;
  g0[0] = 1u;                                               // count=1, user mode
  g0[1] = lds;                                              // lds_addr
  g0[2] = (uint32_t)(ga & 0xffffffffu);                     // global_addr[31:0]
  g0[3] = (uint32_t)((ga >> 32) & 0x01ffffffu) | (2u << 30);// addr[56:32] | type=2
  i32x8 g1;
  g1[0] = (int)(3u << 16);                                  // data_size = 8B
  g1[1] = (int)((tdx & 0xffffu) << 16);                     // tensor_dim0[15:0]
  g1[2] = (int)((tdx >> 16) | ((rows & 0xffffu) << 16));    // tdim0 hi | tensor_dim1 lo
  g1[3] = (int)(((rows >> 16) & 0xffffu) | ((tdx & 0xffffu) << 16)); // tdim1 hi | tile_dim0
  g1[4] = (int)(rows & 0xffffu);                            // tile_dim1 (tile_dim2=0)
  g1[5] = (int)str;                                         // tensor_dim0_stride[31:0]
  g1[6] = 0;                                                // stride hi | dim1 stride lo
  g1[7] = 0;
  i32x4 g2 = {0, 0, 0, 0};
  i32x4 g3 = {0, 0, 0, 0};
  i32x8 gx = {0, 0, 0, 0, 0, 0, 0, 0};
  // clang-23 / therock-10.0 form: 6 args (g0, g1, g2, g3, g4, cpol)
  __builtin_amdgcn_tensor_load_to_lds(g0, g1, g2, g3, gx, 0);
}

// ---------------------------------------------------------------- converts
__global__ __launch_bounds__(256) void f32_to_f16_kernel(
    const float* __restrict__ s, half_t* __restrict__ d, int n) {
  int i = blockIdx.x * 256 + threadIdx.x;
  if (i < n) d[i] = (half_t)s[i];
}

// ---------------------------------------------------------------- WMMA GEMM (double-buffered)
// C[M,N] = alpha * A[M,K] @ B[K,N] (+ beta * add[M,N]); A,B f16 row-major.
// Block tile 128x128, 8 waves (4x2), wave tile 32x64, K-step 32, 2-deep pipeline.
__global__ __launch_bounds__(256) void gemm_f16_kernel(
    const half_t* __restrict__ A, const half_t* __restrict__ Bm, float* __restrict__ C,
    int M, int N, int K, float alpha, const float* __restrict__ addbuf, float beta) {
  __shared__ half_t As[2][128][32];   // [buf][row][k]
  __shared__ half_t Bt[2][128][32];   // [buf][col][k] (transposed at store)
  const int tid  = threadIdx.x;
  const int lane = tid & 31, wid = tid >> 5;
  const int ln   = lane & 15, hi = lane >> 4;
  const int wr   = wid & 3,  wc = wid >> 2;
  const int row0 = blockIdx.y * 128, col0 = blockIdx.x * 128;

  Facc acc[2][4];
#pragma unroll
  for (int i = 0; i < 2; ++i)
#pragma unroll
    for (int c = 0; c < 4; ++c)
#pragma unroll
      for (int r = 0; r < 8; ++r) acc[i][c].f[r] = 0.f;

  const int KT = K >> 5;
  // ---- prologue: async-stage A tile 0; global-load B tile 0 into registers
#pragma unroll
  for (int u = 0; u < 2; ++u) {
    int c = tid + u * 256;
    int r = c >> 2, off = (c & 3) * 8;
    async_g2lds_b128(A + (size_t)(row0 + r) * K + off, &As[0][r][off]);
  }
  half_t breg[2][8];
#pragma unroll
  for (int u = 0; u < 2; ++u) {
    int c = tid + u * 256;
    int kk = c >> 4, off = (c & 15) * 8;
    const half_t* src = Bm + (size_t)kk * N + col0 + off;
#pragma unroll
    for (int j = 0; j < 8; ++j) breg[u][j] = src[j];
  }

  for (int kt = 0; kt < KT; ++kt) {
    const int buf = kt & 1, nbuf = buf ^ 1;
    // store this tile's B registers transposed into LDS
#pragma unroll
    for (int u = 0; u < 2; ++u) {
      int c = tid + u * 256;
      int kk = c >> 4, off = (c & 15) * 8;
#pragma unroll
      for (int j = 0; j < 8; ++j) Bt[buf][off + j][kk] = breg[u][j];
    }
    // prefetch next tile: async A into other buffer, B into registers
    if (kt + 1 < KT) {
      const int k0n = (kt + 1) << 5;
#pragma unroll
      for (int u = 0; u < 2; ++u) {
        int c = tid + u * 256;
        int r = c >> 2, off = (c & 3) * 8;
        async_g2lds_b128(A + (size_t)(row0 + r) * K + k0n + off, &As[nbuf][r][off]);
      }
#pragma unroll
      for (int u = 0; u < 2; ++u) {
        int c = tid + u * 256;
        int kk = c >> 4, off = (c & 15) * 8;
        const half_t* src = Bm + (size_t)(k0n + kk) * N + col0 + off;
#pragma unroll
        for (int j = 0; j < 8; ++j) breg[u][j] = src[j];
      }
      wait_async_le2();   // 2 newest in flight belong to tile kt+1; tile kt landed
    } else {
      wait_async0();
    }
    __syncthreads();

    F16x16 af[2], bf[4];
#pragma unroll
    for (int i = 0; i < 2; ++i) {
      const half_t* ap = &As[buf][wr * 32 + i * 16 + ln][hi * 8];
#pragma unroll
      for (int j = 0; j < 8; ++j) { af[i].h[j] = ap[j]; af[i].h[8 + j] = ap[16 + j]; }
    }
#pragma unroll
    for (int c = 0; c < 4; ++c) {
      const half_t* bp = &Bt[buf][wc * 64 + c * 16 + ln][hi * 8];
#pragma unroll
      for (int j = 0; j < 8; ++j) { bf[c].h[j] = bp[j]; bf[c].h[8 + j] = bp[16 + j]; }
    }
#pragma unroll
    for (int i = 0; i < 2; ++i)
#pragma unroll
      for (int c = 0; c < 4; ++c)
        acc[i][c].v = wmma_f16(af[i].v, bf[c].v, acc[i][c].v);
    __syncthreads();
  }

#pragma unroll
  for (int i = 0; i < 2; ++i)
#pragma unroll
    for (int c = 0; c < 4; ++c)
#pragma unroll
      for (int r = 0; r < 8; ++r) {
        int row = row0 + wr * 32 + i * 16 + r + hi * 8;
        int col = col0 + wc * 64 + c * 16 + ln;
        size_t idx = (size_t)row * N + col;
        float val = alpha * acc[i][c].f[r];
        if (addbuf) val += beta * addbuf[idx];
        C[idx] = val;
      }
}

// ---------------------------------------------------------------- row softmax (D=128)
__global__ __launch_bounds__(128) void softmax128_kernel(
    const float* __restrict__ in, float* __restrict__ out) {
  __shared__ float red[128];
  const int tid = threadIdx.x;
  const size_t row = blockIdx.x;
  float x = in[row * 128 + tid];
  red[tid] = x;
  __syncthreads();
  for (int s = 64; s > 0; s >>= 1) { if (tid < s) red[tid] = fmaxf(red[tid], red[tid + s]); __syncthreads(); }
  float mx = red[0];
  __syncthreads();
  float e = __expf(x - mx);
  red[tid] = e;
  __syncthreads();
  for (int s = 64; s > 0; s >>= 1) { if (tid < s) red[tid] += red[tid + s]; __syncthreads(); }
  out[row * 128 + tid] = e / red[0];
}

// ---------------------------------------------------------------- gate: logsigmoid(k@Wg+bg)/16
__global__ __launch_bounds__(128) void gate128_kernel(
    const float* __restrict__ Kf, const float* __restrict__ Wg,
    const float* __restrict__ bg, float* __restrict__ g) {
  __shared__ float red[128];
  const int tid = threadIdx.x;
  const size_t row = blockIdx.x;
  red[tid] = Kf[row * 128 + tid] * Wg[tid];
  __syncthreads();
  for (int s = 64; s > 0; s >>= 1) { if (tid < s) red[tid] += red[tid + s]; __syncthreads(); }
  if (tid == 0) {
    float z  = red[0] + bg[0];
    float ls = fminf(z, 0.f) - log1pf(__expf(-fabsf(z)));
    g[row] = ls * 0.0625f;
  }
}

// ---------------------------------------------------------------- delta-rule scan
__global__ __launch_bounds__(128) void delta_rule_kernel(
    const float* __restrict__ qs, const float* __restrict__ ks,
    const float* __restrict__ v,  const float* __restrict__ g,
    float* __restrict__ fla) {
  const int b = blockIdx.x >> 4, h = blockIdx.x & 15, kv = h >> 1;
  const int tid = threadIdx.x;
  __shared__ float kt[128], qt[128];
  float S[128];
#pragma unroll
  for (int i = 0; i < 128; ++i) S[i] = 0.f;

  for (int n = 0; n < N_; ++n) {
    size_t qrow = ((size_t)(b * N_ + n) * H_ + h) * D_;
    size_t krow = ((size_t)(b * N_ + n) * HKV_ + kv) * D_;
    qt[tid] = qs[qrow + tid];
    kt[tid] = ks[krow + tid];
    float vtl = v[krow + tid];
    float dec = __expf(g[(size_t)(b * N_ + n) * HKV_ + kv]);
    __syncthreads();
    float dot_k = 0.f;
#pragma unroll
    for (int i = 0; i < 128; ++i) { S[i] *= dec; dot_k += kt[i] * S[i]; }
    float vnew = vtl - dot_k;
    float dot_q = 0.f;
#pragma unroll
    for (int i = 0; i < 128; ++i) { S[i] += kt[i] * vnew; dot_q += qt[i] * S[i]; }
    fla[(size_t)(b * N_ + n) * (H_ * D_) + h * D_ + tid] = SCALE_ * dot_q;
    __syncthreads();
  }
}

// ---------------------------------------------------------------- causal flash attention (WMMA)
__global__ __launch_bounds__(256) void attn_kernel(
    const half_t* __restrict__ Qh, const half_t* __restrict__ Kh,
    const half_t* __restrict__ Vh, float* __restrict__ base) {
  __shared__ half_t Qs[128][128];
  __shared__ half_t Ksm[32][128];
  __shared__ half_t Vtm[128][32];
  __shared__ half_t Pb[8][16][32];
  const int tid = threadIdx.x;
  const int lane = tid & 31, wid = tid >> 5;
  const int ln = lane & 15, hi = lane >> 4;
  const int rb = blockIdx.x, h = blockIdx.y, b = blockIdx.z;
  const int kv = h >> 1;
  const int r0 = rb * 128;
  const int rw = r0 + wid * 16;

  // stage Q rows once: 8 async 16B chunks per thread (ASYNCcnt)
#pragma unroll
  for (int u = 0; u < 8; ++u) {
    int c = tid + u * 256;
    int r = c >> 4, off = (c & 15) * 8;
    async_g2lds_b128(Qh + ((size_t)(b * N_ + r0 + r) * H_ + h) * D_ + off, &Qs[r][off]);
  }

  Facc O[8];
  float mstat[8], lstat[8];
#pragma unroll
  for (int dt = 0; dt < 8; ++dt)
#pragma unroll
    for (int r = 0; r < 8; ++r) O[dt].f[r] = 0.f;
#pragma unroll
  for (int r = 0; r < 8; ++r) { mstat[r] = -1e30f; lstat[r] = 0.f; }

  const int jbmax = (r0 + 127) >> 5;
  for (int jb = 0; jb <= jbmax; ++jb) {
    __syncthreads();
    // K tile (32 tokens x 256B, row stride 2048B) via Tensor Data Mover (wave 0)
    if (wid == 0) {
      tdm_load_2d(&Ksm[0][0],
                  Kh + ((size_t)(b * N_ + jb * 32) * HKV_ + kv) * D_,
                  D_ * 2, 32, HKV_ * D_ * 2);
    }
    // stage V transposed [dim][tok] (VGPR transpose path)
#pragma unroll
    for (int u = 0; u < 16; ++u) {
      int c = tid + u * 256;
      int tok = c >> 7, dim = c & 127;
      Vtm[dim][tok] = Vh[((size_t)(b * N_ + jb * 32 + tok) * HKV_ + kv) * D_ + dim];
    }
    if (wid == 0) __builtin_amdgcn_s_wait_tensorcnt(0);
    if (jb == 0) wait_async0();     // Q tile landed
    __syncthreads();

    const int c0 = jb * 32;
    if (c0 <= rw + 15) {
      float Sv[2][8];
#pragma unroll
      for (int ts = 0; ts < 2; ++ts) {
        Facc sc;
#pragma unroll
        for (int r = 0; r < 8; ++r) sc.f[r] = 0.f;
#pragma unroll
        for (int ks = 0; ks < 4; ++ks) {
          F16x16 a, bb;
          const half_t* ap = &Qs[wid * 16 + ln][ks * 32 + hi * 8];
#pragma unroll
          for (int j = 0; j < 8; ++j) { a.h[j] = ap[j]; a.h[8 + j] = ap[16 + j]; }
          const half_t* bp = &Ksm[ts * 16 + ln][ks * 32 + hi * 8];
#pragma unroll
          for (int j = 0; j < 8; ++j) { bb.h[j] = bp[j]; bb.h[8 + j] = bp[16 + j]; }
          sc.v = wmma_f16(a.v, bb.v, sc.v);
        }
#pragma unroll
        for (int r = 0; r < 8; ++r) {
          int rowg = rw + r + hi * 8;
          int col  = c0 + ts * 16 + ln;
          Sv[ts][r] = (col <= rowg) ? sc.f[r] * SCALE_ : -1e30f;
        }
      }
      float mcur[8];
#pragma unroll
      for (int r = 0; r < 8; ++r) mcur[r] = fmaxf(Sv[0][r], Sv[1][r]);
#pragma unroll
      for (int off = 1; off < 16; off <<= 1)
#pragma unroll
        for (int r = 0; r < 8; ++r) mcur[r] = fmaxf(mcur[r], __shfl_xor(mcur[r], off, 32));
      float alp[8], pv[2][8], psum[8];
#pragma unroll
      for (int r = 0; r < 8; ++r) {
        float mn = fmaxf(mstat[r], mcur[r]);
        alp[r] = __expf(mstat[r] - mn);
        mstat[r] = mn;
        int rowg = rw + r + hi * 8;
#pragma unroll
        for (int ts = 0; ts < 2; ++ts) {
          int col = c0 + ts * 16 + ln;
          pv[ts][r] = (col <= rowg) ? __expf(Sv[ts][r] - mn) : 0.f;
        }
        psum[r] = pv[0][r] + pv[1][r];
      }
#pragma unroll
      for (int off = 1; off < 16; off <<= 1)
#pragma unroll
        for (int r = 0; r < 8; ++r) psum[r] += __shfl_xor(psum[r], off, 32);
#pragma unroll
      for (int r = 0; r < 8; ++r) lstat[r] = lstat[r] * alp[r] + psum[r];
#pragma unroll
      for (int dt = 0; dt < 8; ++dt)
#pragma unroll
        for (int r = 0; r < 8; ++r) O[dt].f[r] *= alp[r];

#pragma unroll
      for (int ts = 0; ts < 2; ++ts)
#pragma unroll
        for (int r = 0; r < 8; ++r)
          Pb[wid][r + hi * 8][ts * 16 + ln] = (half_t)pv[ts][r];
      F16x16 pa;
      const half_t* pp = &Pb[wid][ln][hi * 8];
#pragma unroll
      for (int j = 0; j < 8; ++j) { pa.h[j] = pp[j]; pa.h[8 + j] = pp[16 + j]; }
#pragma unroll
      for (int dt = 0; dt < 8; ++dt) {
        F16x16 vb;
        const half_t* vp = &Vtm[dt * 16 + ln][hi * 8];
#pragma unroll
        for (int j = 0; j < 8; ++j) { vb.h[j] = vp[j]; vb.h[8 + j] = vp[16 + j]; }
        O[dt].v = wmma_f16(pa.v, vb.v, O[dt].v);
      }
    }
  }

#pragma unroll
  for (int dt = 0; dt < 8; ++dt)
#pragma unroll
    for (int r = 0; r < 8; ++r) {
      int rowg = rw + r + hi * 8;
      int col  = dt * 16 + ln;
      base[(size_t)(b * N_ + rowg) * (H_ * D_) + h * D_ + col] = O[dt].f[r] / lstat[r];
    }
}

// ---------------------------------------------------------------- launch
extern "C" void kernel_launch(void* const* d_in, const int* in_sizes, int n_in,
                              void* d_out, int out_size, void* d_ws, size_t ws_size,
                              hipStream_t stream) {
  (void)in_sizes; (void)n_in; (void)out_size; (void)ws_size;
  const float* X  = (const float*)d_in[0];
  const float* Wq = (const float*)d_in[1];
  const float* Wk = (const float*)d_in[2];
  const float* Wv = (const float*)d_in[3];
  const float* Wo = (const float*)d_in[4];
  const float* Wg = (const float*)d_in[5];
  const float* bg = (const float*)d_in[6];
  float* out = (float*)d_out;

  const int M = B_ * N_;
  size_t off = 0;
  auto carve = [&](size_t bytes) -> void* {
    void* p = (char*)d_ws + off;
    off += (bytes + 255) & ~(size_t)255;
    return p;
  };
  half_t* Xh    = (half_t*)carve((size_t)M * HID_ * 2);
  half_t* Wqh   = (half_t*)carve((size_t)HID_ * (H_ * D_) * 2);
  half_t* Wkh   = (half_t*)carve((size_t)HID_ * (HKV_ * D_) * 2);
  half_t* Wvh   = (half_t*)carve((size_t)HID_ * (HKV_ * D_) * 2);
  half_t* Woh   = (half_t*)carve((size_t)(H_ * D_) * HID_ * 2);
  float*  Qf    = (float*)carve((size_t)M * (H_ * D_) * 4);
  float*  Kf    = (float*)carve((size_t)M * (HKV_ * D_) * 4);
  float*  Vf    = (float*)carve((size_t)M * (HKV_ * D_) * 4);
  half_t* Qh    = (half_t*)carve((size_t)M * (H_ * D_) * 2);
  half_t* Kh    = (half_t*)carve((size_t)M * (HKV_ * D_) * 2);
  half_t* Vh    = (half_t*)carve((size_t)M * (HKV_ * D_) * 2);
  float*  qsB   = (float*)carve((size_t)M * (H_ * D_) * 4);
  float*  ksB   = (float*)carve((size_t)M * (HKV_ * D_) * 4);
  float*  gB    = (float*)carve((size_t)M * HKV_ * 4);
  float*  fla   = (float*)carve((size_t)M * (H_ * D_) * 4);
  float*  base  = (float*)carve((size_t)M * (H_ * D_) * 4);
  half_t* baseh = (half_t*)carve((size_t)M * (H_ * D_) * 2);

  auto cvt = [&](const float* s, half_t* d, int n) {
    f32_to_f16_kernel<<<(n + 255) / 256, 256, 0, stream>>>(s, d, n);
  };

  cvt(X,  Xh,  M * HID_);
  cvt(Wq, Wqh, HID_ * H_ * D_);
  cvt(Wk, Wkh, HID_ * HKV_ * D_);
  cvt(Wv, Wvh, HID_ * HKV_ * D_);
  cvt(Wo, Woh, H_ * D_ * HID_);

  gemm_f16_kernel<<<dim3((H_ * D_) / 128, M / 128), 256, 0, stream>>>(
      Xh, Wqh, Qf, M, H_ * D_, HID_, 1.f, nullptr, 0.f);
  gemm_f16_kernel<<<dim3((HKV_ * D_) / 128, M / 128), 256, 0, stream>>>(
      Xh, Wkh, Kf, M, HKV_ * D_, HID_, 1.f, nullptr, 0.f);
  gemm_f16_kernel<<<dim3((HKV_ * D_) / 128, M / 128), 256, 0, stream>>>(
      Xh, Wvh, Vf, M, HKV_ * D_, HID_, 1.f, nullptr, 0.f);

  cvt(Qf, Qh, M * H_ * D_);
  cvt(Kf, Kh, M * HKV_ * D_);
  cvt(Vf, Vh, M * HKV_ * D_);

  softmax128_kernel<<<M * H_,   128, 0, stream>>>(Qf, qsB);
  softmax128_kernel<<<M * HKV_, 128, 0, stream>>>(Kf, ksB);
  gate128_kernel<<<M * HKV_, 128, 0, stream>>>(Kf, Wg, bg, gB);

  delta_rule_kernel<<<B_ * H_, 128, 0, stream>>>(qsB, ksB, Vf, gB, fla);

  attn_kernel<<<dim3(N_ / 128, H_, B_), 256, 0, stream>>>(Qh, Kh, Vh, base);

  cvt(base, baseh, M * H_ * D_);
  gemm_f16_kernel<<<dim3(HID_ / 128, M / 128), 256, 0, stream>>>(
      baseh, Woh, out, M, HID_, H_ * D_, 1.0f - 0.01f, fla, 0.01f);
}
